// DHBR_23716809409204
// MI455X (gfx1250) — matrix operand order repaired and checked by reading, not compiled, not run on hardware
//
// DHBR bundle-recommendation pipeline for MI455X (gfx1250, wave32, WMMA).
// Dense GEMMs use v_wmma_f32_16x16x32_f16 with pre-packed B fragments
// (contiguous 32B-per-lane loads) and register-resident A strips.
#include <hip/hip_runtime.h>
#include <cstdint>
#include <cstddef>

typedef __attribute__((ext_vector_type(16))) _Float16 v16h;
typedef __attribute__((ext_vector_type(8)))  float    v8f;

static __device__ __forceinline__ float lrelu(float x){ return x > 0.f ? x : 0.05f * x; }

static __device__ __forceinline__ void atomAddF(float* p, float v){
  __hip_atomic_fetch_add(p, v, __ATOMIC_RELAXED, __HIP_MEMORY_SCOPE_AGENT);
}

static __device__ __forceinline__ v8f wmma16(v16h a, v16h b, v8f c){
  // 8 args: (neg_a, A, neg_b, B, c_mod, C, reuse_a, reuse_b)
  return __builtin_amdgcn_wmma_f32_16x16x32_f16(false, a, false, b, (short)0, c, false, false);
}

// ---- WMMA fragment loaders (wave32, 16x16x32 f16) ------------------------
// A fragment from row-major MxK f16 source. ISA layout: lane l, m=m0+(l&15),
// halves e<8 -> K = k0 + hi*8 + e ; e>=8 -> K = k0 + 16 + hi*8 + (e-8),
// i.e. two contiguous 16-byte chunks per lane.
static __device__ __forceinline__ v16h frag_a(const _Float16* __restrict__ A, int lda,
                                              int m0, int k0, int lane){
  const int m  = m0 + (lane & 15);
  const int hi = (lane >> 4) & 1;
  const _Float16* p = A + (size_t)m * lda + k0 + hi * 8;
  v16h f;
#pragma unroll
  for (int e = 0; e < 8; ++e){ f[e] = p[e]; f[e + 8] = p[16 + e]; }
  return f;
}

// Packed-B fragment: one contiguous 32-byte load per lane.
// Packed layout (see k_pack_b): tile (kt,nt) holds 512 halves at
// ((kt*NT + nt)*32 + lane)*16 + e   where element e of lane l is
// B[kt*32 + (l>>4)*8 + (e&7) + (e>>3)*16][nt*16 + (l&15)].
static __device__ __forceinline__ v16h frag_bp(const _Float16* __restrict__ P, int NT,
                                               int kt, int nt, int lane){
  const _Float16* p = P + ((size_t)(kt * NT + nt) * 32 + lane) * 16;
  return *(const v16h*)p;
}

// Pack a row-major KxN f32 matrix into WMMA-B fragment order (f16).
__global__ void k_pack_b(const float* __restrict__ src, _Float16* __restrict__ dst,
                         int K, int N){
  int o = blockIdx.x * blockDim.x + threadIdx.x;
  if (o >= K * N) return;
  const int e    = o & 15;
  const int lane = (o >> 4) & 31;
  const int tile = o >> 9;
  const int NT   = N >> 4;
  const int kt   = tile / NT, nt = tile - kt * NT;
  const int n = nt * 16 + (lane & 15);
  const int k = kt * 32 + ((lane >> 4) << 3) + (e & 7) + ((e >> 3) << 4);
  dst[o] = (_Float16)src[(size_t)k * N + n];
}

// ---- elementwise / graph kernels -----------------------------------------
__global__ void k_cvt_f16(const float* __restrict__ in, _Float16* __restrict__ out, int n){
  int i = blockIdx.x * blockDim.x + threadIdx.x;
  if (i < n) out[i] = (_Float16)in[i];
}

__global__ void k_init_cur_acc(const float* __restrict__ fa, int na64,
                               const float* __restrict__ fb, int nb64,
                               float* __restrict__ cur, float* __restrict__ acc){
  int i = blockIdx.x * blockDim.x + threadIdx.x;
  if (i >= na64 + nb64) return;
  float v = (i < na64) ? fa[i] : fb[i - na64];
  cur[i] = v; acc[i] = v;
}

// out[rows[e]] += vals[e] * x[cols[e]]   (D=64 features per node)
__global__ void k_spmm(const int* __restrict__ rows, const int* __restrict__ cols,
                       const float* __restrict__ vals, const float* __restrict__ x,
                       float* __restrict__ out, int ne){
  int e = blockIdx.x * blockDim.x + threadIdx.x;
  if (e >= ne) return;
  const int r = rows[e], c = cols[e];
  const float v = vals[e];
  const float* xr = x + (size_t)c * 64;
  // Prefetch the gather row for the neighboring edge (global_prefetch_b8,
  // speculative: translation faults are dropped silently).
  if (e + 256 < ne) __builtin_prefetch(x + (size_t)cols[e + 256] * 64, 0, 0);
  float* orow = out + (size_t)r * 64;
#pragma unroll
  for (int d = 0; d < 64; ++d) atomAddF(&orow[d], v * xr[d]);
}

// H1 = relu(X @ W1 + b1)  X: na x 64 (f16 row-major), W1p: packed 64x128,
// H1: na x 128 (f16). One wave per 16-row strip; A held in registers.
__global__ void k_gemm_xw1(const _Float16* __restrict__ X, const _Float16* __restrict__ W1p,
                           const float* __restrict__ b1, _Float16* __restrict__ H1){
  const int lane = threadIdx.x;
  const int m0 = blockIdx.x * 16;
  const v16h a0 = frag_a(X, 64, m0, 0,  lane);
  const v16h a1 = frag_a(X, 64, m0, 32, lane);
  const int nlo = lane & 15;
  const int mb  = m0 + 8 * (lane >> 4);
#pragma unroll
  for (int nt = 0; nt < 8; ++nt){
    v8f c = {};
    c = wmma16(a0, frag_bp(W1p, 8, 0, nt, lane), c);
    c = wmma16(a1, frag_bp(W1p, 8, 1, nt, lane), c);
    const int n = nt * 16 + nlo;
    const float bias = b1[n];
#pragma unroll
    for (int r = 0; r < 8; ++r){
      float v = c[r] + bias;
      H1[(size_t)(mb + r) * 128 + n] = (_Float16)(v > 0.f ? v : 0.f);
    }
  }
}

// Z = H1 @ W2   H1: na x 128 (f16 row-major), W2p: packed 128x128, Z: na x 128 (f32)
__global__ void k_gemm_h1w2(const _Float16* __restrict__ H1, const _Float16* __restrict__ W2p,
                            float* __restrict__ Z){
  const int lane = threadIdx.x;
  const int m0 = blockIdx.x * 16;
  v16h a[4];
#pragma unroll
  for (int kt = 0; kt < 4; ++kt) a[kt] = frag_a(H1, 128, m0, kt * 32, lane);
  const int nlo = lane & 15;
  const int mb  = m0 + 8 * (lane >> 4);
#pragma unroll
  for (int nt = 0; nt < 8; ++nt){
    v8f c = {};
#pragma unroll
    for (int kt = 0; kt < 4; ++kt)
      c = wmma16(a[kt], frag_bp(W2p, 8, kt, nt, lane), c);
    const int n = nt * 16 + nlo;
#pragma unroll
    for (int r = 0; r < 8; ++r) Z[(size_t)(mb + r) * 128 + n] = c[r];
  }
}

// row-wise softmax over H=128; writes row-major adj (GEMM4 A operand) and
// transposed adjT (128 x na, GEMM3 A operand).
__global__ void k_softmax128(const float* __restrict__ Z, _Float16* __restrict__ Adj,
                             _Float16* __restrict__ AdjT, int na){
  __shared__ float sm[128];
  const int row = blockIdx.x;
  const int t = threadIdx.x;
  float v = Z[(size_t)row * 128 + t];
  sm[t] = v; __syncthreads();
  for (int s = 64; s > 0; s >>= 1){ if (t < s) sm[t] = fmaxf(sm[t], sm[t + s]); __syncthreads(); }
  const float mx = sm[0]; __syncthreads();
  const float e = __expf(v - mx);
  sm[t] = e; __syncthreads();
  for (int s = 64; s > 0; s >>= 1){ if (t < s) sm[t] += sm[t + s]; __syncthreads(); }
  const _Float16 h = (_Float16)(e * (1.f / sm[0]));
  Adj[(size_t)row * 128 + t] = h;
  AdjT[(size_t)t * na + row] = h;
}

// F1 += Adj^T @ Feat  (128 x 64, split-K over na, atomic reduction).
// AdjT: 128 x na row-major f16; FeatP: packed (na x 64) B fragments.
// block: 256 threads = 8 waves; wave w owns h-tile w, loops 4 n-tiles.
__global__ void k_gemm_adjT_feat(const _Float16* __restrict__ AdjT,
                                 const _Float16* __restrict__ FeatP,
                                 float* __restrict__ F1, int na, int kc){
  const int lane = threadIdx.x & 31;
  const int m0 = (threadIdx.x >> 5) * 16;
  const int kbeg = blockIdx.x * kc;
  int kend = kbeg + kc; if (kend > na) kend = na;   // na is a multiple of 32
  v8f c0 = {}, c1 = {}, c2 = {}, c3 = {};
  for (int k0 = kbeg; k0 < kend; k0 += 32){
    const v16h a = frag_a(AdjT, na, m0, k0, lane);
    const int kt = k0 >> 5;
    c0 = wmma16(a, frag_bp(FeatP, 4, kt, 0, lane), c0);
    c1 = wmma16(a, frag_bp(FeatP, 4, kt, 1, lane), c1);
    c2 = wmma16(a, frag_bp(FeatP, 4, kt, 2, lane), c2);
    c3 = wmma16(a, frag_bp(FeatP, 4, kt, 3, lane), c3);
  }
  const int n  = lane & 15;
  const int mb = m0 + 8 * (lane >> 4);
  v8f cc[4] = { c0, c1, c2, c3 };
#pragma unroll
  for (int nt = 0; nt < 4; ++nt)
#pragma unroll
    for (int r = 0; r < 8; ++r)
      atomAddF(&F1[(size_t)(mb + r) * 64 + nt * 16 + n], cc[nt][r]);
}

__global__ void k_lrelu(float* __restrict__ x, int n){
  int i = blockIdx.x * blockDim.x + threadIdx.x;
  if (i < n) x[i] = lrelu(x[i]);
}

// fout = lrelu(hw @ fin) + fin    (128x128 @ 128x64 — tiny, plain FMA)
__global__ void k_hyper_step(const float* __restrict__ hw, const float* __restrict__ fin,
                             float* __restrict__ fout){
  int i = blockIdx.x * blockDim.x + threadIdx.x;   // 8192 threads
  const int h = i >> 6, d = i & 63;
  float s = 0.f;
#pragma unroll 8
  for (int k = 0; k < 128; ++k) s += hw[h * 128 + k] * fin[k * 64 + d];
  fout[i] = lrelu(s) + fin[i];
}

// curPart += lrelu(Adj @ F4)  Adj: na x 128 (f16 row-major), F4p: packed 128x64.
__global__ void k_gemm_adj_f4(const _Float16* __restrict__ Adj, const _Float16* __restrict__ F4p,
                              float* __restrict__ curPart){
  const int lane = threadIdx.x;
  const int m0 = blockIdx.x * 16;
  v16h a[4];
#pragma unroll
  for (int kt = 0; kt < 4; ++kt) a[kt] = frag_a(Adj, 128, m0, kt * 32, lane);
  const int nlo = lane & 15;
  const int mb  = m0 + 8 * (lane >> 4);
#pragma unroll
  for (int nt = 0; nt < 4; ++nt){
    v8f c = {};
#pragma unroll
    for (int kt = 0; kt < 4; ++kt)
      c = wmma16(a[kt], frag_bp(F4p, 4, kt, nt, lane), c);
    const int n = nt * 16 + nlo;
#pragma unroll
    for (int r = 0; r < 8; ++r)
      curPart[(size_t)(mb + r) * 64 + n] += lrelu(c[r]);
  }
}

// acc[row] += cur[row] / (||cur[row]||2 + 1e-8)
__global__ void k_norm_acc(float* __restrict__ acc, const float* __restrict__ cur, int n){
  int row = blockIdx.x * blockDim.x + threadIdx.x;
  if (row >= n) return;
  const float* cr = cur + (size_t)row * 64;
  float ss = 0.f;
#pragma unroll
  for (int d = 0; d < 64; ++d) ss += cr[d] * cr[d];
  const float inv = 1.f / (sqrtf(ss) + 1e-8f);
  float* ar = acc + (size_t)row * 64;
#pragma unroll
  for (int d = 0; d < 64; ++d) ar[d] += cr[d] * inv;
}

__global__ void k_final_users(const float* __restrict__ aIL, const float* __restrict__ aBL,
                              float* __restrict__ out, int n){
  int i = blockIdx.x * blockDim.x + threadIdx.x;
  if (i < n) out[i] = (aIL[i] + aBL[i]) * (1.f / 3.f);
}

__global__ void k_final_bundles(const float* __restrict__ aBL, float* __restrict__ out,
                                int U64, int n){
  int i = blockIdx.x * blockDim.x + threadIdx.x;
  if (i < n) out[U64 + i] = aBL[U64 + i] * (1.f / 3.f);
}

// out[U + agg_rows[e]] += agg_vals[e] * (accIL[U + agg_cols[e]] / 3)
__global__ void k_agg(const int* __restrict__ r, const int* __restrict__ c,
                      const float* __restrict__ v, const float* __restrict__ aIL,
                      float* __restrict__ out, int ne, int U){
  int e = blockIdx.x * blockDim.x + threadIdx.x;
  if (e >= ne) return;
  const float w = v[e] * (1.f / 3.f);
  const float* src = aIL + (size_t)(U + c[e]) * 64;
  float* dst = out + (size_t)(U + r[e]) * 64;
#pragma unroll
  for (int d = 0; d < 64; ++d) atomAddF(&dst[d], w * src[d]);
}

// ---- host orchestration ---------------------------------------------------
struct LevelScratch {
  float* cur; float* tmp; float* accOut;
  _Float16* feat16; _Float16* featP; _Float16* h116; _Float16* adj16; _Float16* adjT16;
  float* z;
  float* f1; float* f2; float* f3; float* f4;
  _Float16* f4p;
};

static void process_part(hipStream_t s, float* part, int np,
                         const _Float16* w1p, const float* b1, const _Float16* w2p,
                         const float* hw, LevelScratch& ws){
  k_cvt_f16<<<(np * 64 + 255) / 256, 256, 0, s>>>(part, ws.feat16, np * 64);
  k_pack_b<<<(np * 64 + 255) / 256, 256, 0, s>>>(part, ws.featP, np, 64);
  k_gemm_xw1<<<np / 16, 32, 0, s>>>(ws.feat16, w1p, b1, ws.h116);
  k_gemm_h1w2<<<np / 16, 32, 0, s>>>(ws.h116, w2p, ws.z);
  k_softmax128<<<np, 128, 0, s>>>(ws.z, ws.adj16, ws.adjT16, np);
  hipMemsetAsync(ws.f1, 0, 128 * 64 * sizeof(float), s);
  const int KC = 2048;
  k_gemm_adjT_feat<<<(np + KC - 1) / KC, 256, 0, s>>>(ws.adjT16, ws.featP, ws.f1, np, KC);
  k_lrelu<<<32, 256, 0, s>>>(ws.f1, 8192);
  k_hyper_step<<<32, 256, 0, s>>>(hw + 0 * 16384, ws.f1, ws.f2);
  k_hyper_step<<<32, 256, 0, s>>>(hw + 1 * 16384, ws.f2, ws.f3);
  k_hyper_step<<<32, 256, 0, s>>>(hw + 2 * 16384, ws.f3, ws.f4);
  k_pack_b<<<32, 256, 0, s>>>(ws.f4, ws.f4p, 128, 64);
  k_gemm_adj_f4<<<np / 16, 32, 0, s>>>(ws.adj16, ws.f4p, part);
}

static void run_level(hipStream_t s,
                      const float* fa, int na, const float* fb, int nb,
                      const int* rows, const int* cols, const float* vals, int ne,
                      const _Float16* w1a, const float* b1a, const _Float16* w2a,
                      const _Float16* w1b, const float* b1b, const _Float16* w2b,
                      const float* hwa, const float* hwb,
                      LevelScratch& ws, int Lnum){
  const int n = na + nb;
  float* cur = ws.cur;
  float* tmp = ws.tmp;
  float* acc = ws.accOut;
  k_init_cur_acc<<<(n * 64 + 255) / 256, 256, 0, s>>>(fa, na * 64, fb, nb * 64, cur, acc);
  for (int it = 0; it < Lnum; ++it){
    hipMemsetAsync(tmp, 0, (size_t)n * 64 * sizeof(float), s);
    k_spmm<<<(ne + 255) / 256, 256, 0, s>>>(rows, cols, vals, cur, tmp, ne);
    { float* t = cur; cur = tmp; tmp = t; }
    process_part(s, cur, na,
                 w1a + (size_t)it * 64 * 128, b1a + it * 128, w2a + (size_t)it * 128 * 128,
                 hwa, ws);
    process_part(s, cur + (size_t)na * 64, nb,
                 w1b + (size_t)it * 64 * 128, b1b + it * 128, w2b + (size_t)it * 128 * 128,
                 hwb, ws);
    k_norm_acc<<<(n + 255) / 256, 256, 0, s>>>(acc, cur, n);
  }
}

extern "C" void kernel_launch(void* const* d_in, const int* in_sizes, int n_in,
                              void* d_out, int out_size, void* d_ws, size_t ws_size,
                              hipStream_t stream){
  const float* users    = (const float*)d_in[0];
  const float* items    = (const float*)d_in[1];
  const float* bundles  = (const float*)d_in[2];
  const float* il_u_W1  = (const float*)d_in[3];
  const float* il_u_b1  = (const float*)d_in[4];
  const float* il_u_W2  = (const float*)d_in[5];
  const float* il_i_W1  = (const float*)d_in[6];
  const float* il_i_b1  = (const float*)d_in[7];
  const float* il_i_W2  = (const float*)d_in[8];
  const float* bl_u_W1  = (const float*)d_in[9];
  const float* bl_u_b1  = (const float*)d_in[10];
  const float* bl_u_W2  = (const float*)d_in[11];
  const float* bl_b_W1  = (const float*)d_in[12];
  const float* bl_b_b1  = (const float*)d_in[13];
  const float* bl_b_W2  = (const float*)d_in[14];
  const float* il_hw    = (const float*)d_in[15];
  const float* bl_u_hw  = (const float*)d_in[16];
  const float* bl_b_hw  = (const float*)d_in[17];
  const float* il_vals  = (const float*)d_in[18];
  const float* bl_vals  = (const float*)d_in[19];
  const float* agg_vals = (const float*)d_in[20];
  const int*   il_rows  = (const int*)d_in[21];
  const int*   il_cols  = (const int*)d_in[22];
  const int*   bl_rows  = (const int*)d_in[23];
  const int*   bl_cols  = (const int*)d_in[24];
  const int*   agg_rows = (const int*)d_in[25];
  const int*   agg_cols = (const int*)d_in[26];

  const int U  = in_sizes[0] / 64;             // 50000 (multiple of 16)
  const int I  = in_sizes[1] / 64;             // 100000
  const int Bn = in_sizes[2] / 64;             // 20000
  const int Lnum = in_sizes[3] / (64 * 128);   // 2
  const int NIL = U + I, NBL = U + Bn;
  int NA = U; if (I > NA) NA = I; if (Bn > NA) NA = Bn;
  const int ne_il = in_sizes[18], ne_bl = in_sizes[19], ne_agg = in_sizes[20];

  // Carve scratch from d_ws (total ~290 MB).
  char* p = (char*)d_ws;
  auto carve = [&](size_t bytes) -> void* {
    void* r = (void*)p;
    p += (bytes + 255) & ~(size_t)255;
    return r;
  };
  float*    cur    = (float*)   carve((size_t)NIL * 64 * 4);
  float*    tmp    = (float*)   carve((size_t)NIL * 64 * 4);
  float*    accIL  = (float*)   carve((size_t)NIL * 64 * 4);
  float*    accBL  = (float*)   carve((size_t)NBL * 64 * 4);
  _Float16* feat16 = (_Float16*)carve((size_t)NA * 64 * 2);
  _Float16* featP  = (_Float16*)carve((size_t)NA * 64 * 2);
  _Float16* h116   = (_Float16*)carve((size_t)NA * 128 * 2);
  float*    z      = (float*)   carve((size_t)NA * 128 * 4);
  _Float16* adj16  = (_Float16*)carve((size_t)NA * 128 * 2);
  _Float16* adjT16 = (_Float16*)carve((size_t)NA * 128 * 2);
  float*    f1     = (float*)   carve(8192 * 4);
  float*    f2     = (float*)   carve(8192 * 4);
  float*    f3     = (float*)   carve(8192 * 4);
  float*    f4     = (float*)   carve(8192 * 4);
  _Float16* f4p    = (_Float16*)carve(8192 * 2);

  // Pack the 8 GEMM weight tensors (per layer) into B-fragment order.
  const float* wsrc[8] = { il_u_W1, il_u_W2, il_i_W1, il_i_W2,
                           bl_u_W1, bl_u_W2, bl_b_W1, bl_b_W2 };
  _Float16* wpk[8];
  for (int i = 0; i < 8; ++i){
    const bool isW1 = (i & 1) == 0;
    const int  K = isW1 ? 64 : 128, N = 128;
    const size_t per = (size_t)K * N;
    wpk[i] = (_Float16*)carve(per * Lnum * 2);
    for (int l = 0; l < Lnum; ++l)
      k_pack_b<<<(int)((per + 255) / 256), 256, 0, stream>>>(wsrc[i] + per * l,
                                                             wpk[i] + per * l, K, N);
  }

  LevelScratch ws{ cur, tmp, accIL, feat16, featP, h116, adj16, adjT16,
                   z, f1, f2, f3, f4, f4p };

  // Item-level propagation: nodes = users ++ items, hw shared.
  run_level(stream, users, U, items, I, il_rows, il_cols, il_vals, ne_il,
            wpk[0], il_u_b1, wpk[1], wpk[2], il_i_b1, wpk[3],
            il_hw, il_hw, ws, Lnum);

  // Bundle-level propagation: nodes = users ++ bundles.
  ws.accOut = accBL;
  run_level(stream, users, U, bundles, Bn, bl_rows, bl_cols, bl_vals, ne_bl,
            wpk[4], bl_u_b1, wpk[5], wpk[6], bl_b_b1, wpk[7],
            bl_u_hw, bl_b_hw, ws, Lnum);

  float* out = (float*)d_out;
  k_final_users<<<(U * 64 + 255) / 256, 256, 0, stream>>>(accIL, accBL, out, U * 64);
  k_final_bundles<<<(Bn * 64 + 255) / 256, 256, 0, stream>>>(accBL, out, U * 64, Bn * 64);
  k_agg<<<(ne_agg + 255) / 256, 256, 0, stream>>>(agg_rows, agg_cols, agg_vals, accIL,
                                                  out, ne_agg, U);
}
